// LinearSelfAttention_38989713113188
// MI455X (gfx1250) — matrix operand order repaired and checked by reading, not compiled, not run on hardware
//
#include <hip/hip_runtime.h>
#include <hip/hip_bf16.h>

// ---------------------------------------------------------------------------
// LinearSelfAttention (MobileViT-style) for MI455X / gfx1250, wave32 + WMMA
// + Tensor Data Mover staging of the GEMM weight tiles.
// B=4, C=256, QC=513, H=W=192, PH=PW=2 -> P=4, N=96*96=9216.
// ---------------------------------------------------------------------------

#define B_   4
#define C_   256
#define QC_  513
#define HW_  192
#define NW_  96
#define NN_  9216          // nH*nW
#define PLANE_ (192*192)   // == 36864 == 4*9216

typedef __attribute__((ext_vector_type(16))) __bf16 v16bf;
typedef __attribute__((ext_vector_type(8)))  __bf16 v8bf;
typedef __attribute__((ext_vector_type(8)))  float  v8f;
typedef __attribute__((ext_vector_type(4)))  float  v4f;
typedef __attribute__((ext_vector_type(4)))  unsigned int v4u;
typedef __attribute__((ext_vector_type(8)))  int    v8i;
typedef __attribute__((ext_vector_type(4)))  int    v4i;

#define BM 128
#define BN 64
#define BK 32
#define LDA 36   // fp32 A-tile LDS row stride (32 data + 4 pad dwords, via TDM pad)
#define LDK 40   // bf16 B-tile LDS K-stride
#define KSTEPS (C_ / BK)   // 8

// ---------------------------------------------------------------------------
// TDM: async 2D fp32 tile load Global -> LDS (D# per CDNA5 ISA 8.3/8.4).
// dim0 = contiguous K run, dim1 = rows at stride0 elements.
// LDS pad: +4 dwords after every 32 dwords -> row stride LDA=36 floats.
// 6-arg builtin on this toolchain:
//   (uint32x4 g0, int32x8 g1, int32x4 g2, int32x4 g3, int32x8 g4, i32 cpol)
// G1 is loop-invariant; only G0 (lds_addr, global_addr) varies per K-step.
// ---------------------------------------------------------------------------
__device__ __forceinline__ v8i tdm_make_g1(
    unsigned int tensor_d0, unsigned int tensor_d1,
    unsigned int tile_d0, unsigned int tile_d1,
    unsigned long long stride0_elems) {
    v8i g1;
    // data_size=4B (2), pad_enable, pad_interval=32dw (code 4), pad_amount=4dw (code 3)
    g1[0] = (int)((2u << 16) | (1u << 20) | (4u << 22) | (3u << 25));
    g1[1] = (int)((tensor_d0 & 0xffffu) << 16);                       // dim0 lo
    g1[2] = (int)(((tensor_d0 >> 16) & 0xffffu) | ((tensor_d1 & 0xffffu) << 16));
    g1[3] = (int)(((tensor_d1 >> 16) & 0xffffu) | ((tile_d0 & 0xffffu) << 16));
    g1[4] = (int)(tile_d1 & 0xffffu);                                 // tile_dim1, tile_dim2=0
    g1[5] = (int)(unsigned int)(stride0_elems & 0xffffffffull);       // stride0[31:0]
    g1[6] = (int)(unsigned int)((stride0_elems >> 32) & 0xffffull);   // stride0[47:32]
    g1[7] = 0;
    return g1;
}

__device__ __forceinline__ void tdm_issue(
    unsigned int lds_byte_addr, const float* gaddr, v8i g1) {
    const unsigned long long ga = (unsigned long long)(uintptr_t)gaddr;
    v4u g0;
    g0[0] = 1u;                                     // count=1, user desc
    g0[1] = lds_byte_addr;                          // lds_addr
    g0[2] = (unsigned int)(ga & 0xffffffffu);       // global_addr[31:0]
    g0[3] = (unsigned int)((ga >> 32) & 0x01ffffffu) | (2u << 30);  // type=2
    const v4i g2 = {0, 0, 0, 0};
    const v4i g3 = {0, 0, 0, 0};
    const v8i g4 = {0, 0, 0, 0, 0, 0, 0, 0};
    __builtin_amdgcn_tensor_load_to_lds(g0, g1, g2, g3, g4, 0);
}

// Depthwise 3x3 "SAME" conv over a (4, 9216) image stored row-major.
__device__ __forceinline__ float dwconv_at(const float* __restrict__ plane,
                                           const float wk[9], int p, int n) {
    float s = 0.f;
    #pragma unroll
    for (int dp = -1; dp <= 1; ++dp) {
        int pp = p + dp;
        if (pp < 0 || pp > 3) continue;
        const float* row = plane + pp * NN_;
        #pragma unroll
        for (int dn = -1; dn <= 1; ++dn) {
            int nn = n + dn;
            if (nn < 0 || nn >= NN_) continue;
            s += wk[(dp + 1) * 3 + (dn + 1)] * row[nn];
        }
    }
    return s;
}

// Build a v16bf A-fragment (contiguous k = kb + s, matching the B fragment) from an fp32 LDS row.
__device__ __forceinline__ v16bf a_frag_f32(const float* __restrict__ row, int kb) {
    const v4f q0 = *(const v4f*)(row + kb);
    const v4f q1 = *(const v4f*)(row + kb + 4);
    const v4f q2 = *(const v4f*)(row + kb + 8);
    const v4f q3 = *(const v4f*)(row + kb + 12);
    v16bf a;
    #pragma unroll
    for (int i = 0; i < 4; ++i) {
        a[i]      = (__bf16)q0[i];
        a[4 + i]  = (__bf16)q1[i];
        a[8 + i]  = (__bf16)q2[i];
        a[12 + i] = (__bf16)q3[i];
    }
    return a;
}

// ---------------------------------------------------------------------------
// Kernel 1: qkv projection GEMM (TDM-staged A, 128x64 tile, 4 WMMA/wave/step)
//   qkv[b, o, p, n] = sum_c patches[b,c,p,n] * qkv_w[o,c] + qkv_b[o]
// ---------------------------------------------------------------------------
__global__ __launch_bounds__(256) void qkv_gemm(
    const float* __restrict__ x, const float* __restrict__ qkv_w,
    const float* __restrict__ qkv_b, float* __restrict__ qkv) {

    __shared__ __align__(16) float  sAf[2][BM * LDA];  // fp32, filled by TDM
    __shared__ __align__(16) __bf16 sB[BN * LDK];      // bf16 [col][k]

    const int t = threadIdx.x;
    const int bz = blockIdx.z;
    const int b = bz >> 2, p = bz & 3;
    const int py = p >> 1, px = p & 1;
    const int nbase = blockIdx.x * BN;
    const int mbase = blockIdx.y * BM;

    const int wave = t >> 5, lane = t & 31;
    const int wm = wave >> 1;            // 0..3 -> 32-row group
    const int wn = wave & 1;             // 0..1 -> 32-col group
    const int lhalf = lane >> 4, l15 = lane & 15;
    const int kb = 16 * lhalf;

    v8f acc00 = {}, acc01 = {}, acc10 = {}, acc11 = {};

    // B staging: thread -> (col = t&63, k-group = t>>6): 8 consecutive k
    const int b_j = t & 63, b_kg = t >> 6;
    const int ncol = nbase + b_j;
    const int nh = ncol / NW_, nw = ncol - nh * NW_;
    const int gy = 2 * nh + py, gx = 2 * nw + px;
    const long long xoff = (long long)b * C_ * PLANE_ + (long long)gy * HW_ + gx;

    const unsigned int ldsA[2] = {(unsigned int)(uintptr_t)&sAf[0][0],
                                  (unsigned int)(uintptr_t)&sAf[1][0]};
    const float* abase = qkv_w + (long long)mbase * C_;
    // Loop-invariant D# group 1; edge rows (>= QC_) read as zeros.
    const v8i g1 = tdm_make_g1(C_, (unsigned int)(QC_ - mbase), BK, BM, C_);

    // Prologue: TDM prefetch of A tile for k0 = 0 into buffer 0 (wave 0 only).
    if (t < 32) tdm_issue(ldsA[0], abase, g1);

    #pragma unroll
    for (int i = 0; i < KSTEPS; ++i) {
        const int k0 = i * BK;
        const int cur = i & 1;

        // Stage B tile 32x64 (patch gather, fp32 -> bf16, transposed [col][k]).
        {
            const float* src = x + xoff + (long long)(k0 + b_kg * 8) * PLANE_;
            if (i + 1 < KSTEPS) __builtin_prefetch(src + (long long)BK * PLANE_, 0, 3);
            __bf16* dst = &sB[b_j * LDK + b_kg * 8];
            #pragma unroll
            for (int j = 0; j < 8; ++j) dst[j] = (__bf16)src[(long long)j * PLANE_];
        }

        __builtin_amdgcn_s_wait_tensorcnt(0);  // A tile (cur) complete (wave0)
        __syncthreads();

        // Kick next A tile into the other buffer while we compute.
        if (i + 1 < KSTEPS && t < 32)
            tdm_issue(ldsA[cur ^ 1], abase + (k0 + BK), g1);

        // Fragments + 4 WMMAs.
        const float* arow0 = &sAf[cur][(wm * 32 + l15) * LDA];
        const float* arow1 = &sAf[cur][(wm * 32 + 16 + l15) * LDA];
        const v16bf a0 = a_frag_f32(arow0, kb);
        const v16bf a1 = a_frag_f32(arow1, kb);
        const v16bf b0 = *(const v16bf*)&sB[(wn * 32 + l15) * LDK + 16 * lhalf];
        const v16bf b1 = *(const v16bf*)&sB[(wn * 32 + 16 + l15) * LDK + 16 * lhalf];

        acc00 = __builtin_amdgcn_wmma_f32_16x16x32_bf16(false, a0, false, b0, (short)0, acc00, false, false);
        acc01 = __builtin_amdgcn_wmma_f32_16x16x32_bf16(false, a0, false, b1, (short)0, acc01, false, false);
        acc10 = __builtin_amdgcn_wmma_f32_16x16x32_bf16(false, a1, false, b0, (short)0, acc10, false, false);
        acc11 = __builtin_amdgcn_wmma_f32_16x16x32_bf16(false, a1, false, b1, (short)0, acc11, false, false);
        __syncthreads();
    }

    // Store: lane -> col = l15; VGPR r -> row = 8*lhalf + r (per 16-row sub-tile)
    const int col0 = nbase + wn * 32 + l15;
    const int col1 = col0 + 16;
    #pragma unroll
    for (int sub = 0; sub < 2; ++sub) {
        const int rowb = mbase + wm * 32 + sub * 16 + 8 * lhalf;
        const v8f acL = sub ? acc10 : acc00;
        const v8f acR = sub ? acc11 : acc01;
        #pragma unroll
        for (int r = 0; r < 8; ++r) {
            const int o = rowb + r;
            if (o < QC_) {
                const float bias = qkv_b[o];
                const long long base = (((long long)b * QC_ + o) * 4 + p) * NN_;
                qkv[base + col0] = acL[r] + bias;
                qkv[base + col1] = acR[r] + bias;
            }
        }
    }
}

// ---------------------------------------------------------------------------
// Kernel 2: softmax over n of the conv'd query channel. 16 blocks = (b,p).
// ---------------------------------------------------------------------------
__global__ __launch_bounds__(256) void softmax_query(
    const float* __restrict__ qkv, const float* __restrict__ dw_w,
    const float* __restrict__ dw_b, float* __restrict__ scores) {

    __shared__ float red[256];
    const int bz = blockIdx.x, b = bz >> 2, p = bz & 3;
    const float* qplane = qkv + ((long long)b * QC_) * 4 * NN_;  // channel 0
    float wk[9];
    #pragma unroll
    for (int i = 0; i < 9; ++i) wk[i] = dw_w[i];
    const float bias = dw_b[0];
    float* srow = scores + (long long)bz * NN_;

    float mx = -1e30f;
    for (int n = threadIdx.x; n < NN_; n += 256) {
        const float v = dwconv_at(qplane, wk, p, n) + bias;
        srow[n] = v;
        mx = fmaxf(mx, v);
    }
    red[threadIdx.x] = mx; __syncthreads();
    for (int s = 128; s > 0; s >>= 1) {
        if (threadIdx.x < s) red[threadIdx.x] = fmaxf(red[threadIdx.x], red[threadIdx.x + s]);
        __syncthreads();
    }
    mx = red[0]; __syncthreads();

    float sum = 0.f;
    for (int n = threadIdx.x; n < NN_; n += 256) {
        const float e = __expf(srow[n] - mx);
        srow[n] = e; sum += e;
    }
    red[threadIdx.x] = sum; __syncthreads();
    for (int s = 128; s > 0; s >>= 1) {
        if (threadIdx.x < s) red[threadIdx.x] += red[threadIdx.x + s];
        __syncthreads();
    }
    const float inv = 1.f / red[0];
    for (int n = threadIdx.x; n < NN_; n += 256) srow[n] *= inv;
}

// ---------------------------------------------------------------------------
// Kernel 3: context[b,c,p] = sum_n conv(key_c)(p,n) * scores[b,p,n] + dw_b.
// ---------------------------------------------------------------------------
__global__ __launch_bounds__(256) void context_kernel(
    const float* __restrict__ qkv, const float* __restrict__ dw_w,
    const float* __restrict__ dw_b, const float* __restrict__ scores,
    float* __restrict__ context) {

    __shared__ float red[256];
    const int id = blockIdx.x;
    const int p = id & 3, c = (id >> 2) & 255, b = id >> 10;
    const int ch = 1 + c;
    const float* kplane = qkv + (((long long)b * QC_ + ch) * 4) * NN_;
    const float* srow = scores + (long long)(b * 4 + p) * NN_;
    float wk[9];
    #pragma unroll
    for (int i = 0; i < 9; ++i) wk[i] = dw_w[ch * 9 + i];

    float sum = 0.f;
    for (int n = threadIdx.x; n < NN_; n += 256)
        sum += dwconv_at(kplane, wk, p, n) * srow[n];

    red[threadIdx.x] = sum; __syncthreads();
    for (int s = 128; s > 0; s >>= 1) {
        if (threadIdx.x < s) red[threadIdx.x] += red[threadIdx.x + s];
        __syncthreads();
    }
    if (threadIdx.x == 0) context[id] = red[0] + dw_b[ch];  // sum(scores)==1
}

// ---------------------------------------------------------------------------
// Kernel 4: vprep = relu(conv(value)+bias) * context, into dead key slots.
// ---------------------------------------------------------------------------
__global__ __launch_bounds__(256) void vprep_kernel(
    const float* __restrict__ qkv_ro, const float* __restrict__ dw_w,
    const float* __restrict__ dw_b, const float* __restrict__ context,
    float* __restrict__ qkv_wr) {

    const long long idx = (long long)blockIdx.x * 256 + threadIdx.x;
    const int n = (int)(idx % NN_);
    long long r = idx / NN_;
    const int p = (int)(r & 3);
    r >>= 2;
    const int c = (int)(r & 255);
    const int b = (int)(r >> 8);
    const int ch = 1 + C_ + c;  // value channel

    const float* vplane = qkv_ro + (((long long)b * QC_ + ch) * 4) * NN_;
    float wk[9];
    #pragma unroll
    for (int i = 0; i < 9; ++i) wk[i] = dw_w[ch * 9 + i];

    float v = dwconv_at(vplane, wk, p, n) + dw_b[ch];
    v = fmaxf(v, 0.f);
    v *= context[((b * C_) + c) * 4 + p];
    qkv_wr[(((long long)b * QC_ + (1 + c)) * 4 + p) * NN_ + n] = v;
}

// ---------------------------------------------------------------------------
// Kernel 5: proj GEMM + fold (TDM-staged A, 128x64 tile).
//   out[b,o,2nh+py,2nw+px] = sum_c proj_w[o,c]*vprep[b,c,p,n] + proj_b[o]
// ---------------------------------------------------------------------------
__global__ __launch_bounds__(256) void proj_gemm(
    const float* __restrict__ qkv /* vprep in key slots */,
    const float* __restrict__ proj_w, const float* __restrict__ proj_b,
    float* __restrict__ out) {

    __shared__ __align__(16) float  sAf[2][BM * LDA];
    __shared__ __align__(16) __bf16 sB[BN * LDK];

    const int t = threadIdx.x;
    const int bz = blockIdx.z;
    const int b = bz >> 2, p = bz & 3;
    const int py = p >> 1, px = p & 1;
    const int nbase = blockIdx.x * BN;
    const int mbase = blockIdx.y * BM;

    const int wave = t >> 5, lane = t & 31;
    const int wm = wave >> 1, wn = wave & 1;
    const int lhalf = lane >> 4, l15 = lane & 15;
    const int kb = 16 * lhalf;

    v8f acc00 = {}, acc01 = {}, acc10 = {}, acc11 = {};

    const int b_j = t & 63, b_kg = t >> 6;
    const int ncol = nbase + b_j;
    const float* vbase = qkv + (((long long)b * QC_ + 1) * 4 + p) * NN_ + ncol;

    const unsigned int ldsA[2] = {(unsigned int)(uintptr_t)&sAf[0][0],
                                  (unsigned int)(uintptr_t)&sAf[1][0]};
    const float* abase = proj_w + (long long)mbase * C_;
    const v8i g1 = tdm_make_g1(C_, (unsigned int)(C_ - mbase), BK, BM, C_);

    if (t < 32) tdm_issue(ldsA[0], abase, g1);

    #pragma unroll
    for (int i = 0; i < KSTEPS; ++i) {
        const int k0 = i * BK;
        const int cur = i & 1;

        {   // B tile 32x64: vprep rows, fully coalesced along n.
            const float* src = vbase + (long long)(k0 + b_kg * 8) * (4 * NN_);
            if (i + 1 < KSTEPS) __builtin_prefetch(src + (long long)BK * 4 * NN_, 0, 3);
            __bf16* dst = &sB[b_j * LDK + b_kg * 8];
            #pragma unroll
            for (int j = 0; j < 8; ++j) dst[j] = (__bf16)src[(long long)j * (4 * NN_)];
        }

        __builtin_amdgcn_s_wait_tensorcnt(0);
        __syncthreads();

        if (i + 1 < KSTEPS && t < 32)
            tdm_issue(ldsA[cur ^ 1], abase + (k0 + BK), g1);

        const float* arow0 = &sAf[cur][(wm * 32 + l15) * LDA];
        const float* arow1 = &sAf[cur][(wm * 32 + 16 + l15) * LDA];
        const v16bf a0 = a_frag_f32(arow0, kb);
        const v16bf a1 = a_frag_f32(arow1, kb);
        const v16bf b0 = *(const v16bf*)&sB[(wn * 32 + l15) * LDK + 16 * lhalf];
        const v16bf b1 = *(const v16bf*)&sB[(wn * 32 + 16 + l15) * LDK + 16 * lhalf];

        acc00 = __builtin_amdgcn_wmma_f32_16x16x32_bf16(false, a0, false, b0, (short)0, acc00, false, false);
        acc01 = __builtin_amdgcn_wmma_f32_16x16x32_bf16(false, a0, false, b1, (short)0, acc01, false, false);
        acc10 = __builtin_amdgcn_wmma_f32_16x16x32_bf16(false, a1, false, b0, (short)0, acc10, false, false);
        acc11 = __builtin_amdgcn_wmma_f32_16x16x32_bf16(false, a1, false, b1, (short)0, acc11, false, false);
        __syncthreads();
    }

    const int col0 = nbase + wn * 32 + l15;
    const int col1 = col0 + 16;
    const int nh0 = col0 / NW_, nw0 = col0 - nh0 * NW_;
    const int nh1 = col1 / NW_, nw1 = col1 - nh1 * NW_;
    const long long o0 = (long long)(2 * nh0 + py) * HW_ + (2 * nw0 + px);
    const long long o1 = (long long)(2 * nh1 + py) * HW_ + (2 * nw1 + px);

    #pragma unroll
    for (int sub = 0; sub < 2; ++sub) {
        const int rowb = mbase + wm * 32 + sub * 16 + 8 * lhalf;
        const v8f acL = sub ? acc10 : acc00;
        const v8f acR = sub ? acc11 : acc01;
        #pragma unroll
        for (int r = 0; r < 8; ++r) {
            const int o = rowb + r;
            const float bias = proj_b[o];
            float* oplane = out + (long long)(b * C_ + o) * PLANE_;
            oplane[o0] = acL[r] + bias;
            oplane[o1] = acR[r] + bias;
        }
    }
}

// ---------------------------------------------------------------------------
extern "C" void kernel_launch(void* const* d_in, const int* in_sizes, int n_in,
                              void* d_out, int out_size, void* d_ws, size_t ws_size,
                              hipStream_t stream) {
    const float* x      = (const float*)d_in[0];
    const float* qkv_w  = (const float*)d_in[1];
    const float* qkv_b  = (const float*)d_in[2];
    const float* dw_w   = (const float*)d_in[3];
    const float* dw_b   = (const float*)d_in[4];
    const float* proj_w = (const float*)d_in[5];
    const float* proj_b = (const float*)d_in[6];
    float* out = (float*)d_out;

    float* qkv     = (float*)d_ws;                        // 4*513*4*9216 floats
    float* scores  = qkv + (long long)B_ * QC_ * 4 * NN_; // 4*4*9216 floats
    float* context = scores + (long long)B_ * 4 * NN_;    // 4*256*4 floats

    const dim3 blk(256);
    qkv_gemm      <<<dim3(NN_ / BN, (QC_ + BM - 1) / BM, B_ * 4), blk, 0, stream>>>(x, qkv_w, qkv_b, qkv);
    softmax_query <<<dim3(B_ * 4),                                blk, 0, stream>>>(qkv, dw_w, dw_b, scores);
    context_kernel<<<dim3(B_ * C_ * 4),                           blk, 0, stream>>>(qkv, dw_w, dw_b, scores, context);
    vprep_kernel  <<<dim3((B_ * C_ * 4 * NN_) / 256),             blk, 0, stream>>>(qkv, dw_w, dw_b, context, qkv);
    proj_gemm     <<<dim3(NN_ / BN, C_ / BM, B_ * 4),             blk, 0, stream>>>(qkv, proj_w, proj_b, out);
}
